// CFGNN_75651553952189
// MI455X (gfx1250) — matrix operand order repaired and checked by reading, not compiled, not run on hardware
//
#include <hip/hip_runtime.h>

// ---------------------------------------------------------------------------
// CFGNN for MI455X (gfx1250, wave32, WMMA bf16 16x16x32)
//
// Sizes: B=32, N=96, T=16, H=256, D=512, V=20002, E=160
// ---------------------------------------------------------------------------

typedef __attribute__((ext_vector_type(16))) __bf16 v16bf;
typedef __attribute__((ext_vector_type(8)))  float  v8f;

#define BB   32
#define NN   96
#define TT   16
#define HH   256
#define DD   512
#define EE   160
#define MROWS (BB*NN)     // 3072

// ------------------------- WMMA helpers ------------------------------------

__device__ __forceinline__ v16bf ld_frag2(const __bf16* p0, const __bf16* p1) {
  v16bf v;
  reinterpret_cast<uint4*>(&v)[0] = *reinterpret_cast<const uint4*>(p0);
  reinterpret_cast<uint4*>(&v)[1] = *reinterpret_cast<const uint4*>(p1);
  return v;
}

// A fragment: 16x32 bf16 tile, rows row0..row0+15, K block starting at k0.
// ISA layout: lanes 0-15 -> K=[k0..k0+7] then [k0+16..k0+23];
//             lanes 16-31 -> K=[k0+8..k0+15] then [k0+24..k0+31].
__device__ __forceinline__ v16bf a_frag(const __bf16* A, int lda, int row0,
                                        int k0, int lane) {
  int r = lane & 15, hi = lane >> 4;
  const __bf16* p = A + (size_t)(row0 + r) * lda + k0 + hi * 8;
  return ld_frag2(p, p + 16);
}

// B fragment from a pre-transposed weight Bt[N][K]:
// lane holds column col0+(lane&15); elements = K = k0 + (lane>>4)*8 + 0..7, then +16
__device__ __forceinline__ v16bf b_frag(const __bf16* Bt, int ldb, int col0,
                                        int k0, int lane) {
  int r = lane & 15, hi = lane >> 4;
  const __bf16* p = Bt + (size_t)(col0 + r) * ldb + k0 + hi * 8;
  return ld_frag2(p, p + 16);
}

__device__ __forceinline__ v8f wmma_bf16(v16bf a, v16bf b, v8f c) {
  return __builtin_amdgcn_wmma_f32_16x16x32_bf16(false, a, false, b,
                                                 (short)0, c, false, false);
}

__device__ __forceinline__ float sigmoidf_(float x) {
  return 1.f / (1.f + __expf(-x));
}

// ------------------------- prep kernels ------------------------------------

// xp = mean_t emb[x]  (emb row 1 forced to zero), also bf16 mirror
__global__ __launch_bounds__(256) void k_embed(const int* __restrict__ x,
                                               const float* __restrict__ emb,
                                               float* __restrict__ xp,
                                               __bf16* __restrict__ xpb) {
  int node = blockIdx.x;          // b*96+n
  int h = threadIdx.x;            // 0..255
  float acc = 0.f;
#pragma unroll
  for (int t = 0; t < TT; ++t) {
    int tok = x[node * TT + t];
    if (tok != 1) acc += emb[(size_t)tok * HH + h];
  }
  acc *= (1.f / 16.f);
  xp[(size_t)node * HH + h] = acc;
  xpb[(size_t)node * HH + h] = (__bf16)acc;
}

__device__ __forceinline__ int wrapm(int v) {
  int m = v % NN;
  return m < 0 ? m + NN : m;
}

// Build the 4 adjacency bitmask tables: adj[b][row] = 96-bit mask of cols.
__global__ void k_adj(const int* __restrict__ edges, unsigned* F1,
                      unsigned* B1, unsigned* B2, unsigned* F2) {
  int idx = blockIdx.x * blockDim.x + threadIdx.x;  // b*E+e
  if (idx >= BB * EE) return;
  int b = idx / EE, e = idx % EE;
  const int* fe = edges + ((size_t)((0 * BB + b) * EE + e)) * 2;
  const int* be = edges + ((size_t)((1 * BB + b) * EE + e)) * 2;
  int fs = fe[0], fd = fe[1];
  int bs = be[0], bd = be[1];
  // f1 = to_matrix(f_edges, off=1): mat[b, d-1, s-1] = 1
  { int r = wrapm(fd - 1), c = wrapm(fs - 1);
    atomicOr(&F1[(b * NN + r) * 4 + (c >> 5)], 1u << (c & 31)); }
  // b2 = transpose(to_matrix(f_edges, off=2)): row = s-2, col = d-2
  { int r = wrapm(fs - 2), c = wrapm(fd - 2);
    atomicOr(&B2[(b * NN + r) * 4 + (c >> 5)], 1u << (c & 31)); }
  // b1 = to_matrix(b_edges, off=1)
  { int r = wrapm(bd - 1), c = wrapm(bs - 1);
    atomicOr(&B1[(b * NN + r) * 4 + (c >> 5)], 1u << (c & 31)); }
  // f2 = transpose(to_matrix(b_edges, off=2))
  { int r = wrapm(bs - 2), c = wrapm(bd - 2);
    atomicOr(&F2[(b * NN + r) * 4 + (c >> 5)], 1u << (c & 31)); }
}

// mat[:,0,0] = 0 fix (applied pre-transpose; [0][0] is transpose-invariant)
__global__ void k_adj_fix(unsigned* F1, unsigned* B1, unsigned* B2,
                          unsigned* F2) {
  int b = threadIdx.x;
  if (b < BB) {
    F1[(b * NN) * 4] &= ~1u;
    B1[(b * NN) * 4] &= ~1u;
    B2[(b * NN) * 4] &= ~1u;
    F2[(b * NN) * 4] &= ~1u;
  }
}

// Pack [Wa (256x768) | Wf (256x256)] -> transposed bf16 dst[1024][256]
__global__ __launch_bounds__(256) void k_pack_w(const float* __restrict__ Wa,
                                                const float* __restrict__ Wf,
                                                __bf16* __restrict__ dst) {
  int idx = blockIdx.x * 256 + threadIdx.x;  // 1024*256 total
  int n = idx >> 8, k = idx & 255;
  float v = (n < 768) ? Wa[(size_t)k * 768 + n] : Wf[(size_t)k * 256 + (n - 768)];
  dst[(size_t)n * 256 + k] = (__bf16)v;
}

// bias pre-sum: [bax+bah | bfx+bfh] -> dst[1024]
__global__ void k_pack_bias(const float* ba, const float* bh, const float* bx,
                            const float* bf, float* dst) {
  int n = blockIdx.x * 256 + threadIdx.x;
  if (n < 1024)
    dst[n] = (n < 768) ? (ba[n] + bh[n]) : (bx[n - 768] + bf[n - 768]);
}

// Wc (512x512) -> transposed bf16 (512x512)
__global__ __launch_bounds__(256) void k_tr_cvt(const float* __restrict__ src,
                                                __bf16* __restrict__ dst) {
  int idx = blockIdx.x * 256 + threadIdx.x;  // 512*512
  int n = idx >> 9, k = idx & 511;
  dst[(size_t)n * 512 + k] = (__bf16)src[(size_t)k * 512 + n];
}

// ------------------------- big parallel WMMA GEMM --------------------------
// C[M][Nc] (f32) = A[M][K](bf16) @ Bt[Nc][K](bf16)^T  (+ bias)
// grid = (Nc/128, M/64), block = 256 (8 waves).
// Wave w: m-pair = w&1 (32 rows), n-pair = w>>1 (32 cols) -> 2x2 WMMA tiles,
// software-pipelined over K (next slab's fragments loaded before this slab's
// 4 WMMAs issue, so s_wait only drains loads that are a full iter in flight).
__global__ __launch_bounds__(256) void k_gemm_bf16(
    const __bf16* __restrict__ A, const __bf16* __restrict__ Bt,
    float* __restrict__ C, const float* __restrict__ bias, int M, int Nc,
    int K) {
  int lane = threadIdx.x & 31;
  int w = threadIdx.x >> 5;
  int row0 = blockIdx.y * 64 + (w & 1) * 32;
  int col0 = blockIdx.x * 128 + (w >> 1) * 32;

  v8f acc00 = {0, 0, 0, 0, 0, 0, 0, 0};
  v8f acc01 = acc00, acc10 = acc00, acc11 = acc00;

  v16bf a0 = a_frag(A, K, row0, 0, lane);
  v16bf a1 = a_frag(A, K, row0 + 16, 0, lane);
  v16bf b0 = b_frag(Bt, K, col0, 0, lane);
  v16bf b1 = b_frag(Bt, K, col0 + 16, 0, lane);

  for (int k0 = 0; k0 < K; k0 += 32) {
    int kn = (k0 + 32 < K) ? k0 + 32 : 0;  // clamp: last iter reloads slab 0
    if (k0 + 64 < K) {                     // prefetch two slabs ahead
      __builtin_prefetch(A + (size_t)row0 * K + k0 + 64, 0, 1);
      __builtin_prefetch(Bt + (size_t)col0 * K + k0 + 64, 0, 1);
    }
    v16bf na0 = a_frag(A, K, row0, kn, lane);
    v16bf na1 = a_frag(A, K, row0 + 16, kn, lane);
    v16bf nb0 = b_frag(Bt, K, col0, kn, lane);
    v16bf nb1 = b_frag(Bt, K, col0 + 16, kn, lane);

    acc00 = wmma_bf16(a0, b0, acc00);
    acc01 = wmma_bf16(a0, b1, acc01);
    acc10 = wmma_bf16(a1, b0, acc10);
    acc11 = wmma_bf16(a1, b1, acc11);

    a0 = na0; a1 = na1; b0 = nb0; b1 = nb1;
  }

  int r = lane & 15, hi = lane >> 4;
#pragma unroll
  for (int mt = 0; mt < 2; ++mt)
#pragma unroll
    for (int nt = 0; nt < 2; ++nt) {
      v8f acc = mt ? (nt ? acc11 : acc10) : (nt ? acc01 : acc00);
      int col = col0 + nt * 16 + r;
      float bv = bias ? bias[col] : 0.f;
#pragma unroll
      for (int rr = 0; rr < 8; ++rr) {
        int row = row0 + mt * 16 + rr + hi * 8;
        C[(size_t)row * Nc + col] = acc[rr] + bv;
      }
    }
}

// ------------------------- persistent recurrent scan -----------------------
// One workgroup (8 waves) runs the whole 96-step gated scan.
// Per step:  hl/cl = masked row-sum of state  ->  LDS (bf16 / f32)
//            G[32][1024] = hl @ [Wah|Wfh]   (WMMA, wave w owns cols w*32..+32
//                                            of each of the 4 gate blocks)
//            gate elementwise using precomputed Xpre = xp@[Wax|Wfx]+biases
__global__ __launch_bounds__(256) void k_scan(
    const unsigned* __restrict__ adj,   // [32][96][4] bitmasks
    const float* __restrict__ Xpre,     // [3072][1024]
    const __bf16* __restrict__ Wgt,     // [1024][256] transposed bf16
    float* __restrict__ H, float* __restrict__ Cst,  // [32][96][256]
    int reverse) {
  __shared__ __bf16 hl[BB][HH];  // 16 KB
  __shared__ float cl[BB][HH];   // 32 KB
  int tid = threadIdx.x, lane = tid & 31, w = tid >> 5;
  int r = lane & 15, hi = lane >> 4;

  for (int s = 0; s < NN; ++s) {
    int node = reverse ? (NN - 1 - s) : s;

    // ---- aggregation: hl[b][h] = sum_n adj[b][node][n] * H[b][n][h] ----
    for (int t = tid; t < BB * HH; t += 256) {
      int b = t >> 8, hh = t & 255;
      const unsigned* mrow = adj + (size_t)(b * NN + node) * 4;
      float sh = 0.f, sc = 0.f;
#pragma unroll
      for (int wd = 0; wd < 3; ++wd) {
        unsigned bits = mrow[wd];
        while (bits) {
          int n = wd * 32 + __builtin_ctz(bits);
          bits &= bits - 1;
          size_t idx = ((size_t)(b * NN + n)) * HH + hh;
          sh += H[idx];
          sc += Cst[idx];
        }
      }
      hl[b][hh] = (__bf16)sh;
      cl[b][hh] = sc;
    }
    __syncthreads();

    // ---- WMMA step GEMM: [32 x 256] @ [256 x 1024] ----
    v8f acc[4][2][2];
    v8f zero = {0, 0, 0, 0, 0, 0, 0, 0};
#pragma unroll
    for (int g = 0; g < 4; ++g)
#pragma unroll
      for (int ns = 0; ns < 2; ++ns) {
        acc[g][ns][0] = zero;
        acc[g][ns][1] = zero;
      }
#pragma unroll
    for (int kb = 0; kb < 8; ++kb) {
      int k0 = kb * 32;
      v16bf a0 = a_frag(&hl[0][0], HH, 0, k0, lane);
      v16bf a1 = a_frag(&hl[0][0], HH, 16, k0, lane);
#pragma unroll
      for (int g = 0; g < 4; ++g)
#pragma unroll
        for (int ns = 0; ns < 2; ++ns) {
          int n0 = g * 256 + w * 32 + ns * 16;
          v16bf bf = b_frag(Wgt, HH, n0, k0, lane);
          acc[g][ns][0] = wmma_bf16(a0, bf, acc[g][ns][0]);
          acc[g][ns][1] = wmma_bf16(a1, bf, acc[g][ns][1]);
        }
    }

    // ---- gate elementwise + state write ----
#pragma unroll
    for (int ns = 0; ns < 2; ++ns)
#pragma unroll
      for (int m = 0; m < 2; ++m)
#pragma unroll
        for (int rr = 0; rr < 8; ++rr) {
          int b = m * 16 + rr + hi * 8;        // batch row (0..31)
          int j = w * 32 + ns * 16 + r;        // hidden col (0..255)
          size_t xb = ((size_t)(b * NN + node)) * 1024;
          float iv = acc[0][ns][m][rr] + Xpre[xb + j];
          float ov = acc[1][ns][m][rr] + Xpre[xb + 256 + j];
          float uv = acc[2][ns][m][rr] + Xpre[xb + 512 + j];
          float fv = acc[3][ns][m][rr] + Xpre[xb + 768 + j];
          float cv = sigmoidf_(iv) * tanhf(uv) + sigmoidf_(fv) * cl[b][j];
          float hv = sigmoidf_(ov) * tanhf(cv);
          size_t oi = ((size_t)(b * NN + node)) * HH + j;
          Cst[oi] = cv;
          H[oi] = hv;
        }
    __threadfence();
    __syncthreads();
  }
}

// Sequential in-place aggregation pass: H[:,j,:] += adj-masked row sum, j asc.
__global__ __launch_bounds__(256) void k_aggr_inplace(
    const unsigned* __restrict__ adj, float* __restrict__ H) {
  int tid = threadIdx.x;
  for (int j = 0; j < NN; ++j) {
    for (int t = tid; t < BB * HH; t += 256) {
      int b = t >> 8, hh = t & 255;
      const unsigned* mrow = adj + (size_t)(b * NN + j) * 4;
      float s = 0.f;
#pragma unroll
      for (int wd = 0; wd < 3; ++wd) {
        unsigned bits = mrow[wd];
        while (bits) {
          int n = wd * 32 + __builtin_ctz(bits);
          bits &= bits - 1;
          s += H[((size_t)(b * NN + n)) * HH + hh];
        }
      }
      H[((size_t)(b * NN + j)) * HH + hh] += s;
    }
    __threadfence();
    __syncthreads();
  }
}

// hcat bf16 (cols 0..255 = h, 256..511 = hb) + per-batch max-pool "source"
__global__ __launch_bounds__(256) void k_hcat_source(
    const float* __restrict__ H, const float* __restrict__ HB,
    __bf16* __restrict__ hcat, float* __restrict__ source) {
  int b = blockIdx.x;
  for (int d = threadIdx.x; d < DD; d += 256) {
    const float* src = (d < HH) ? H : HB;
    int dd = d & (HH - 1);
    float mx = -3.0e38f;
    for (int n = 0; n < NN; ++n) {
      float v = src[((size_t)(b * NN + n)) * HH + dd];
      hcat[((size_t)(b * NN + n)) * DD + d] = (__bf16)v;
      mx = fmaxf(mx, v);
    }
    source[(size_t)b * DD + d] = mx;
  }
}

// wq = source @ Wq + bq  (32 x 512 x 512, latency-trivial)
__global__ __launch_bounds__(256) void k_wq(const float* __restrict__ source,
                                            const float* __restrict__ Wq,
                                            const float* __restrict__ bq,
                                            float* __restrict__ wq) {
  __shared__ float s[DD];
  int b = blockIdx.x;
  for (int k = threadIdx.x; k < DD; k += 256) s[k] = source[(size_t)b * DD + k];
  __syncthreads();
  for (int d = threadIdx.x; d < DD; d += 256) {
    float acc = bq[d];
    for (int k = 0; k < DD; ++k) acc += s[k] * Wq[(size_t)k * DD + d];
    wq[(size_t)b * DD + d] = acc;
  }
}

// attention tail: align -> softmax -> context -> fc -> sigmoid
__global__ __launch_bounds__(256) void k_attn(
    const float* __restrict__ uh, const float* __restrict__ wq,
    const float* __restrict__ v, const int* __restrict__ target,
    const float* __restrict__ ann_emb, const float* __restrict__ ann_w,
    const float* __restrict__ ann_b, const float* __restrict__ H,
    const float* __restrict__ HB, const float* __restrict__ source,
    const float* __restrict__ fcW, const float* __restrict__ fcb,
    float* __restrict__ out) {
  __shared__ float align_s[NN];
  __shared__ float ctx_s[DD];
  __shared__ float red[256];
  int b = blockIdx.x, tid = threadIdx.x;

  if (tid < NN) {
    float s = 0.f;
    size_t base = ((size_t)(b * NN + tid)) * DD;
    size_t wb = (size_t)b * DD;
    for (int d = 0; d < DD; ++d)
      s += tanhf(wq[wb + d] + uh[base + d]) * v[d];
    float ann = ann_emb[target[b * NN + tid]] * ann_w[0] + ann_b[0];
    align_s[tid] = s * ann;
  }
  __syncthreads();
  if (tid == 0) {
    float mx = align_s[0];
    for (int n = 1; n < NN; ++n) mx = fmaxf(mx, align_s[n]);
    float sum = 0.f;
    for (int n = 0; n < NN; ++n) {
      float e = __expf(align_s[n] - mx);
      align_s[n] = e;
      sum += e;
    }
    float inv = 1.f / sum;
    for (int n = 0; n < NN; ++n) align_s[n] *= inv;
  }
  __syncthreads();
  for (int d = tid; d < DD; d += 256) {
    const float* src = (d < HH) ? H : HB;
    int dd = d & (HH - 1);
    float s = 0.f;
    for (int n = 0; n < NN; ++n)
      s += align_s[n] * src[((size_t)(b * NN + n)) * HH + dd];
    ctx_s[d] = s;
  }
  __syncthreads();
  float part = 0.f;
  for (int d = tid; d < DD; d += 256)
    part += ctx_s[d] * fcW[d] + source[(size_t)b * DD + d] * fcW[DD + d];
  red[tid] = part;
  __syncthreads();
  for (int off = 128; off; off >>= 1) {
    if (tid < off) red[tid] += red[tid + off];
    __syncthreads();
  }
  if (tid == 0) out[b] = sigmoidf_(red[0] + fcb[0]);
}

// ------------------------- host launcher -----------------------------------

extern "C" void kernel_launch(void* const* d_in, const int* in_sizes, int n_in,
                              void* d_out, int out_size, void* d_ws,
                              size_t ws_size, hipStream_t stream) {
  (void)in_sizes; (void)n_in; (void)out_size; (void)ws_size;

  const int*   x      = (const int*)d_in[0];
  const int*   edges  = (const int*)d_in[1];
  const int*   target = (const int*)d_in[2];
  const float* emb    = (const float*)d_in[3];
  // gate_f
  const float* Wax_f = (const float*)d_in[4];
  const float* bax_f = (const float*)d_in[5];
  const float* Wah_f = (const float*)d_in[6];
  const float* bah_f = (const float*)d_in[7];
  const float* Wfx_f = (const float*)d_in[8];
  const float* bfx_f = (const float*)d_in[9];
  const float* Wfh_f = (const float*)d_in[10];
  const float* bfh_f = (const float*)d_in[11];
  // gate_b
  const float* Wax_b = (const float*)d_in[12];
  const float* bax_b = (const float*)d_in[13];
  const float* Wah_b = (const float*)d_in[14];
  const float* bah_b = (const float*)d_in[15];
  const float* Wfx_b = (const float*)d_in[16];
  const float* bfx_b = (const float*)d_in[17];
  const float* Wfh_b = (const float*)d_in[18];
  const float* bfh_b = (const float*)d_in[19];
  // att
  const float* Wq      = (const float*)d_in[20];
  const float* bq      = (const float*)d_in[21];
  const float* Wc      = (const float*)d_in[22];
  const float* vatt    = (const float*)d_in[23];
  const float* ann_emb = (const float*)d_in[24];
  const float* ann_w   = (const float*)d_in[25];
  const float* ann_b   = (const float*)d_in[26];
  const float* fcW     = (const float*)d_in[27];
  const float* fcb     = (const float*)d_in[28];

  // ---- workspace layout ----
  char* ws = (char*)d_ws;
  size_t off = 0;
  auto alloc = [&](size_t bytes) -> void* {
    void* p = ws + off;
    off += (bytes + 255) & ~(size_t)255;
    return p;
  };
  float*    xp    = (float*)alloc((size_t)MROWS * HH * 4);
  __bf16*   xpb   = (__bf16*)alloc((size_t)MROWS * HH * 2);
  unsigned* F1    = (unsigned*)alloc((size_t)BB * NN * 4 * 4);
  unsigned* B1    = (unsigned*)alloc((size_t)BB * NN * 4 * 4);
  unsigned* B2    = (unsigned*)alloc((size_t)BB * NN * 4 * 4);
  unsigned* F2    = (unsigned*)alloc((size_t)BB * NN * 4 * 4);
  __bf16*   WgtF  = (__bf16*)alloc((size_t)1024 * 256 * 2);
  __bf16*   WgtB  = (__bf16*)alloc((size_t)1024 * 256 * 2);
  __bf16*   WpreF = (__bf16*)alloc((size_t)1024 * 256 * 2);
  __bf16*   WpreB = (__bf16*)alloc((size_t)1024 * 256 * 2);
  __bf16*   WcT   = (__bf16*)alloc((size_t)512 * 512 * 2);
  float*    bpreF = (float*)alloc(1024 * 4);
  float*    bpreB = (float*)alloc(1024 * 4);
  float*    XpreF = (float*)alloc((size_t)MROWS * 1024 * 4);
  float*    XpreB = (float*)alloc((size_t)MROWS * 1024 * 4);
  float*    Hst   = (float*)alloc((size_t)MROWS * HH * 4);
  float*    Cst   = (float*)alloc((size_t)MROWS * HH * 4);
  float*    HBst  = (float*)alloc((size_t)MROWS * HH * 4);
  float*    CBst  = (float*)alloc((size_t)MROWS * HH * 4);
  __bf16*   hcat  = (__bf16*)alloc((size_t)MROWS * DD * 2);
  float*    srcm  = (float*)alloc((size_t)BB * DD * 4);
  float*    uh    = (float*)alloc((size_t)MROWS * DD * 4);
  float*    wqv   = (float*)alloc((size_t)BB * DD * 4);

  // ---- zero state / masks (graph-capturable memsets) ----
  hipMemsetAsync(F1, 0, (size_t)BB * NN * 4 * 4, stream);
  hipMemsetAsync(B1, 0, (size_t)BB * NN * 4 * 4, stream);
  hipMemsetAsync(B2, 0, (size_t)BB * NN * 4 * 4, stream);
  hipMemsetAsync(F2, 0, (size_t)BB * NN * 4 * 4, stream);
  hipMemsetAsync(Hst, 0, (size_t)MROWS * HH * 4, stream);
  hipMemsetAsync(Cst, 0, (size_t)MROWS * HH * 4, stream);
  hipMemsetAsync(HBst, 0, (size_t)MROWS * HH * 4, stream);
  hipMemsetAsync(CBst, 0, (size_t)MROWS * HH * 4, stream);

  // ---- prep ----
  k_embed<<<MROWS, 256, 0, stream>>>(x, emb, xp, xpb);
  k_adj<<<(BB * EE + 255) / 256, 256, 0, stream>>>(edges, F1, B1, B2, F2);
  k_adj_fix<<<1, 32, 0, stream>>>(F1, B1, B2, F2);
  k_pack_w<<<1024, 256, 0, stream>>>(Wah_f, Wfh_f, WgtF);
  k_pack_w<<<1024, 256, 0, stream>>>(Wah_b, Wfh_b, WgtB);
  k_pack_w<<<1024, 256, 0, stream>>>(Wax_f, Wfx_f, WpreF);
  k_pack_w<<<1024, 256, 0, stream>>>(Wax_b, Wfx_b, WpreB);
  k_pack_bias<<<4, 256, 0, stream>>>(bax_f, bah_f, bfx_f, bfh_f, bpreF);
  k_pack_bias<<<4, 256, 0, stream>>>(bax_b, bah_b, bfx_b, bfh_b, bpreB);
  k_tr_cvt<<<1024, 256, 0, stream>>>(Wc, WcT);

  // ---- hoisted input-side GEMMs: Xpre = xp @ [Wax|Wfx] + (bx + bh) ----
  dim3 gpre(1024 / 128, MROWS / 64);
  k_gemm_bf16<<<gpre, 256, 0, stream>>>(xpb, WpreF, XpreF, bpreF, MROWS, 1024, HH);
  k_gemm_bf16<<<gpre, 256, 0, stream>>>(xpb, WpreB, XpreB, bpreB, MROWS, 1024, HH);

  // ---- sequential passes (persistent single-WG kernels) ----
  k_scan<<<1, 256, 0, stream>>>(F1, XpreF, WgtF, Hst, Cst, 0);
  k_aggr_inplace<<<1, 256, 0, stream>>>(B1, Hst);
  k_scan<<<1, 256, 0, stream>>>(B2, XpreB, WgtB, HBst, CBst, 1);
  k_aggr_inplace<<<1, 256, 0, stream>>>(F2, HBst);

  // ---- attention ----
  k_hcat_source<<<BB, 256, 0, stream>>>(Hst, HBst, hcat, srcm);
  dim3 guh(DD / 128, MROWS / 64);
  k_gemm_bf16<<<guh, 256, 0, stream>>>(hcat, WcT, uh, nullptr, MROWS, DD, DD);
  k_wq<<<BB, 256, 0, stream>>>(srcm, Wq, bq, wqv);
  k_attn<<<BB, 256, 0, stream>>>(uh, wqv, vatt, target, ann_emb, ann_w, ann_b,
                                 Hst, HBst, srcm, fcW, fcb, (float*)d_out);
}
